// CausalAttention_18829136625860
// MI455X (gfx1250) — matrix operand order repaired and checked by reading, not compiled, not run on hardware
//
#include <hip/hip_runtime.h>
#include <stdint.h>

#define DI 1024        // D_IN == D_OUT
#define SS 2048        // sequence length
#define NB 4           // batch
#define BSROWS 8192    // NB * SS

#define AS1 __attribute__((address_space(1)))
#define AS3 __attribute__((address_space(3)))

typedef __attribute__((ext_vector_type(16))) __bf16 v16bf;
typedef __attribute__((ext_vector_type(8)))  float  v8f;
typedef __attribute__((ext_vector_type(4)))  int    v4i;
typedef __attribute__((ext_vector_type(8)))  int    v8i;
typedef __attribute__((ext_vector_type(4)))  unsigned int v4u;

union FragBF {
  v16bf v;
  uint4 q[2];
};

// ---- gfx1250 async/TDM feature probes (fallbacks keep compile safe) --------
#if __has_builtin(__builtin_amdgcn_global_load_async_to_lds_b128)
#define HAVE_ASYNC_LDS 1
#endif
#if __has_builtin(__builtin_amdgcn_tensor_load_to_lds)
#define HAVE_TDM 1
#endif

__device__ __forceinline__ unsigned short f32_to_bf16(float f) {
  unsigned int u = __float_as_uint(f);
  u += 0x7FFFu + ((u >> 16) & 1u);   // round-to-nearest-even
  return (unsigned short)(u >> 16);
}

#ifdef HAVE_ASYNC_LDS
__device__ __forceinline__ void async_cp16(void* lds, const void* g) {
  // GLOBAL_LOAD_ASYNC_TO_LDS_B128: 16B DMA, ASYNCcnt-tracked, no VGPR round trip
  // builtin signature: (v4i AS1* gsrc, v4i AS3* ldst, imm offset, imm cpol)
  __builtin_amdgcn_global_load_async_to_lds_b128(
      (AS1 v4i*)(v4i*)g, (AS3 v4i*)(v4i*)lds, 0, 0);
}
#endif

__device__ __forceinline__ void wait_async_all() {
#ifdef HAVE_ASYNC_LDS
#if __has_builtin(__builtin_amdgcn_s_wait_asynccnt)
  __builtin_amdgcn_s_wait_asynccnt(0);
#else
  asm volatile("s_wait_asynccnt 0x0" ::: "memory");
#endif
#endif
}

__device__ __forceinline__ unsigned lds_off_u32(const void* p) {
  // ptrtoint of an AS(3) pointer == wave-relative LDS byte offset
  return (unsigned)(unsigned long long)(AS3 void*)(void*)p;
}

// ---------------------------------------------------------------------------
// Kernel 1: x fp32 -> bf16 (row-major [BSROWS, DI])
// ---------------------------------------------------------------------------
__global__ void cvt_x_bf16(const float* __restrict__ x,
                           unsigned short* __restrict__ xb) {
  size_t i = ((size_t)blockIdx.x * 256 + threadIdx.x) * 8;
  float4 a = *(const float4*)(x + i);
  float4 c = *(const float4*)(x + i + 4);
  uint4 o;
  o.x = (unsigned)f32_to_bf16(a.x) | ((unsigned)f32_to_bf16(a.y) << 16);
  o.y = (unsigned)f32_to_bf16(a.z) | ((unsigned)f32_to_bf16(a.w) << 16);
  o.z = (unsigned)f32_to_bf16(c.x) | ((unsigned)f32_to_bf16(c.y) << 16);
  o.w = (unsigned)f32_to_bf16(c.z) | ((unsigned)f32_to_bf16(c.w) << 16);
  *(uint4*)(xb + i) = o;
}

// ---------------------------------------------------------------------------
// Kernel 2: W fp32 [DI,DI] -> Wt bf16 [DI,DI] transposed (Wt[n][k] = W[k][n])
// ---------------------------------------------------------------------------
__global__ void cvt_w_t(const float* __restrict__ Wq,
                        const float* __restrict__ Wk,
                        const float* __restrict__ Wv,
                        unsigned short* __restrict__ WtAll) {
  __shared__ float tile[32][33];
  const int z = blockIdx.z;
  const float* w = (z == 0) ? Wq : (z == 1) ? Wk : Wv;
  unsigned short* o = WtAll + (size_t)z * DI * DI;
  const int x = blockIdx.x * 32 + threadIdx.x;
  const int y0 = blockIdx.y * 32;
#pragma unroll
  for (int j = 0; j < 4; ++j)
    tile[threadIdx.y + 8 * j][threadIdx.x] =
        w[(size_t)(y0 + threadIdx.y + 8 * j) * DI + x];
  __syncthreads();
  const int n0 = blockIdx.x * 32;
#pragma unroll
  for (int j = 0; j < 4; ++j)
    o[(size_t)(n0 + threadIdx.y + 8 * j) * DI + y0 + threadIdx.x] =
        f32_to_bf16(tile[threadIdx.x][threadIdx.y + 8 * j]);
}

// ---------------------------------------------------------------------------
// Kernel 3: QKV projection GEMM, bf16 WMMA, f32 accumulate.
//   128(M) x 64(N) block tile, K-step 32, double-buffered LDS staged with
//   GLOBAL_LOAD_ASYNC_TO_LDS_B128 (tile i+1 in flight while computing tile i).
//   z==0 -> Q [BSROWS,DI]; z==1 -> K [BSROWS,DI]; z==2 -> Vt [NB,DI,SS]
// ---------------------------------------------------------------------------
__global__ __launch_bounds__(256) void qkv_gemm(
    const unsigned short* __restrict__ xb,
    const unsigned short* __restrict__ wtAll,
    unsigned short* __restrict__ qo,
    unsigned short* __restrict__ ko,
    unsigned short* __restrict__ vto) {
  __shared__ uint4 xT4[2 * 512];   // 2 x (128 rows x 32 k bf16) = 16 KB
  __shared__ uint4 wT4[2 * 256];   // 2 x (64 rows x 32 k bf16)  = 8 KB
  const int tid = threadIdx.x;
  const int wave = tid >> 5, lane = tid & 31;
  const int half = lane >> 4, ln16 = lane & 15;
  const int m0 = blockIdx.y * 128, n0 = blockIdx.x * 64;
  const int z = blockIdx.z;
  const uint4* xg = (const uint4*)xb;
  const uint4* wg = (const uint4*)(wtAll + (size_t)z * DI * DI);

  v8f acc[4] = {};

  auto stage = [&](int buf, int k0) {
    uint4* xd = xT4 + buf * 512;
    uint4* wd = wT4 + buf * 256;
    const int c0 = tid, c1 = tid + 256;
    const int r0 = c0 >> 2, p0 = c0 & 3;
    const int r1 = c1 >> 2, p1 = c1 & 3;
    const int rw = tid >> 2, pw = tid & 3;
#ifdef HAVE_ASYNC_LDS
    async_cp16(&xd[c0], &xg[(size_t)(m0 + r0) * (DI / 8) + (k0 >> 3) + p0]);
    async_cp16(&xd[c1], &xg[(size_t)(m0 + r1) * (DI / 8) + (k0 >> 3) + p1]);
    async_cp16(&wd[tid], &wg[(size_t)(n0 + rw) * (DI / 8) + (k0 >> 3) + pw]);
#else
    uint4 t0 = xg[(size_t)(m0 + r0) * (DI / 8) + (k0 >> 3) + p0];
    uint4 t1 = xg[(size_t)(m0 + r1) * (DI / 8) + (k0 >> 3) + p1];
    uint4 t2 = wg[(size_t)(n0 + rw) * (DI / 8) + (k0 >> 3) + pw];
    xd[c0] = t0;
    xd[c1] = t1;
    wd[tid] = t2;
#endif
  };

  const int NK = DI / 32;   // 32 K-steps
  stage(0, 0);              // preload tile 0

  for (int it = 0; it < NK; ++it) {
    wait_async_all();       // this wave's DMA for tile `it` landed
    __syncthreads();        // ...and everyone else's; prev buffer now free
    if (it + 1 < NK) stage((it + 1) & 1, (it + 1) * 32);

    const uint4* xc = xT4 + (it & 1) * 512;
    const uint4* wc = wT4 + (it & 1) * 256;

    FragBF A;                              // 16x32 A: lane ln16 = row m
    const int m = wave * 16 + ln16;
    A.q[0] = xc[m * 4 + half];             // K = half*8 .. +8
    A.q[1] = xc[m * 4 + 2 + half];         // K = 16+half*8 .. +8
#pragma unroll
    for (int nt = 0; nt < 4; ++nt) {
      FragBF Bf;                           // 32x16 B: lane ln16 = col n
      const int n = nt * 16 + ln16;
      Bf.q[0] = wc[n * 4 + half];
      Bf.q[1] = wc[n * 4 + 2 + half];
      acc[nt] = __builtin_amdgcn_wmma_f32_16x16x32_bf16(
          false, A.v, false, Bf.v, (short)0, acc[nt], false, false);
    }
    __syncthreads();        // all reads of this buffer done before reuse
  }

  // store: C layout = VGPR i: row (i + half*8), col = lane%16
#pragma unroll
  for (int nt = 0; nt < 4; ++nt) {
    const int col = n0 + nt * 16 + ln16;
#pragma unroll
    for (int i = 0; i < 8; ++i) {
      const int g = m0 + wave * 16 + i + half * 8;
      const unsigned short hv = f32_to_bf16(acc[nt][i]);
      if (z == 0)      qo[(size_t)g * DI + col] = hv;
      else if (z == 1) ko[(size_t)g * DI + col] = hv;
      else {
        const int bb = g >> 11, s = g & (SS - 1);
        vto[(size_t)(bb * DI + col) * SS + s] = hv;   // Vt[b][d][s]
      }
    }
  }
}

// ---------------------------------------------------------------------------
// Kernel 4: causal flash attention, one block per (batch, 16 query rows).
//   Q tile (16 x 1024 bf16 = 32 KB) staged by the Tensor Data Mover (one
//   2D-tile descriptor, wave 0 issues, s_wait_tensorcnt completes).
//   8 waves split D=1024 into 128-wide chunks; QK^T partials reduced via
//   ds_add_f32; online softmax; P·V accumulated per-wave. K/Vt stream from
//   global (L2-resident; zero intra-block reuse under the d-split).
// ---------------------------------------------------------------------------
__global__ __launch_bounds__(256) void flash_attn(
    const unsigned short* __restrict__ qg,
    const unsigned short* __restrict__ kg,
    const unsigned short* __restrict__ vtg,
    float* __restrict__ out) {
  __shared__ unsigned short qT[16 * DI];   // 32 KB
  __shared__ float scoreS[16 * 32];
  __shared__ unsigned short pT[16 * 32];
  __shared__ float mrow[16], lrow[16], srow[16];

  const int tid = threadIdx.x;
  const int wave = tid >> 5, lane = tid & 31;
  const int half = lane >> 4, ln16 = lane & 15;
  const int b = blockIdx.y;
  const int q0 = blockIdx.x * 16;

  const uint4* qg4 = (const uint4*)qg;
  uint4* qT4 = (uint4*)qT;

#ifdef HAVE_TDM
  if (wave == 0) {
    // D# group0: count=1 | lds_addr | global_addr[56:0] | type=2
    const unsigned long long ga =
        (unsigned long long)qg + (unsigned long long)(b * SS + q0) * DI * 2ull;
    v4u g0;
    g0.x = 1u;
    g0.y = lds_off_u32(qT);
    g0.z = (unsigned)ga;
    g0.w = (unsigned)((ga >> 32) & 0x01FFFFFFull) | (2u << 30);
    // D# group1: data_size=2B; tensor 1024 x 16; tile 1024 x 16; stride0=1024
    v8i g1;
    g1[0] = (int)(1u << 16);        // data_size=1 (2 bytes), wg_mask=0
    g1[1] = (int)(1024u << 16);     // tensor_dim0[15:0] = 1024
    g1[2] = (int)(16u << 16);       // tensor_dim0 hi=0 | tensor_dim1 lo=16
    g1[3] = (int)(1024u << 16);     // tensor_dim1 hi=0 | tile_dim0=1024
    g1[4] = 16;                     // tile_dim1=16, tile_dim2=0
    g1[5] = 1024;                   // tensor_dim0_stride lo
    g1[6] = 0;                      // stride0 hi | tensor_dim1_stride lo
    g1[7] = 0;
    v4i z4 = {0, 0, 0, 0};
    v8i z8 = {0, 0, 0, 0, 0, 0, 0, 0};
    // 6-arg form on this toolchain: (g0, g1, g2, g3, g?, cpol)
    __builtin_amdgcn_tensor_load_to_lds(g0, g1, z4, z4, z8, 0);
    __builtin_amdgcn_s_wait_tensorcnt(0);
  }
#else
#pragma unroll
  for (int j = 0; j < 8; ++j) {
    int c = tid + 256 * j;
    int row = c >> 7, part = c & 127;
    qT4[c] = qg4[(size_t)(b * SS + q0 + row) * (DI / 8) + part];
  }
#endif
  if (tid < 16) { mrow[tid] = -__builtin_inff(); lrow[tid] = 0.f; }

  v8f acc[8] = {};
  const float inv_sqrt_d = 0.03125f;   // 1/sqrt(1024)
  const int ktmax = (q0 + 15) >> 5;    // inclusive causal bound
  const uint4* kg4 = (const uint4*)kg;
  const uint4* vt4 = (const uint4*)vtg;
  uint4* pT4 = (uint4*)pT;

  for (int kt = 0; kt <= ktmax; ++kt) {
    const int k0 = kt * 32;
    scoreS[tid] = 0.f;
    scoreS[tid + 256] = 0.f;
    __syncthreads();   // also publishes the staged Q tile on iter 0

    // partial scores over this wave's d-chunk [wave*128, wave*128+128)
    v8f sc[2] = {};
#pragma unroll
    for (int ks = 0; ks < 4; ++ks) {
      const int kbase = wave * 128 + ks * 32;
      FragBF A;
      A.q[0] = qT4[ln16 * 128 + (kbase >> 3) + half];
      A.q[1] = qT4[ln16 * 128 + (kbase >> 3) + 2 + half];
#pragma unroll
      for (int nt = 0; nt < 2; ++nt) {
        const int key = k0 + nt * 16 + ln16;
        FragBF Bf;
        const size_t base =
            (size_t)(b * SS + key) * (DI / 8) + (kbase >> 3) + half;
        Bf.q[0] = kg4[base];
        Bf.q[1] = kg4[base + 2];
        sc[nt] = __builtin_amdgcn_wmma_f32_16x16x32_bf16(
            false, A.v, false, Bf.v, (short)0, sc[nt], false, false);
      }
    }
#pragma unroll
    for (int nt = 0; nt < 2; ++nt)
#pragma unroll
      for (int i = 0; i < 8; ++i)
        atomicAdd(&scoreS[(i + half * 8) * 32 + nt * 16 + ln16], sc[nt][i]);

    if (kt < ktmax) {   // prefetch next key tile
      const int keyn = k0 + 32 + (tid & 31);
      __builtin_prefetch(&kg[(size_t)(b * SS + keyn) * DI + ((tid >> 5) << 7)],
                         0, 0);
    }
    __syncthreads();

    // online softmax on the 16x32 score tile (one thread per query row)
    if (tid < 16) {
      const int r = tid;
      const int qrow = q0 + r;
      float sv[32];
      float tmax = -__builtin_inff();
      for (int c = 0; c < 32; ++c) {
        float s = scoreS[r * 32 + c] * inv_sqrt_d;
        if (k0 + c > qrow) s = -__builtin_inff();
        sv[c] = s;
        tmax = fmaxf(tmax, s);
      }
      const float mo = mrow[r];
      const float mn = fmaxf(mo, tmax);
      const float rescale = __expf(mo - mn);
      float sum = 0.f;
      for (int c = 0; c < 32; ++c) {
        const float p = __expf(sv[c] - mn);
        sum += p;
        pT[r * 32 + c] = f32_to_bf16(p);
      }
      mrow[r] = mn;
      lrow[r] = lrow[r] * rescale + sum;
      srow[r] = rescale;
    }
    __syncthreads();

    // rescale running output and accumulate P·V for this wave's d-chunk
    float rs[8];
#pragma unroll
    for (int i = 0; i < 8; ++i) rs[i] = srow[i + half * 8];
    FragBF Ap;   // P: 16 q-rows x 32 keys
    Ap.q[0] = pT4[ln16 * 4 + half];
    Ap.q[1] = pT4[ln16 * 4 + 2 + half];
#pragma unroll
    for (int nt = 0; nt < 8; ++nt) {
#pragma unroll
      for (int i = 0; i < 8; ++i) acc[nt][i] *= rs[i];
      const int d = wave * 128 + nt * 16 + ln16;
      FragBF Bv;   // Vt[b][d][k0 + half*8 .. +8] and [k0 + 16 + half*8 .. +8]
      const size_t base =
          (size_t)(b * DI + d) * (SS / 8) + (k0 >> 3) + half;
      Bv.q[0] = vt4[base];
      Bv.q[1] = vt4[base + 2];
      acc[nt] = __builtin_amdgcn_wmma_f32_16x16x32_bf16(
          false, Ap.v, false, Bv.v, (short)0, acc[nt], false, false);
    }
  }

  // final normalize + fp32 store
  float li[8];
#pragma unroll
  for (int i = 0; i < 8; ++i) li[i] = 1.0f / lrow[i + half * 8];
#pragma unroll
  for (int nt = 0; nt < 8; ++nt) {
    const int d = wave * 128 + nt * 16 + ln16;
#pragma unroll
    for (int i = 0; i < 8; ++i) {
      const int row = q0 + i + half * 8;
      out[(size_t)(b * SS + row) * DI + d] = acc[nt][i] * li[i];
    }
  }
}

// ---------------------------------------------------------------------------
// Host launcher
// ---------------------------------------------------------------------------
extern "C" void kernel_launch(void* const* d_in, const int* in_sizes, int n_in,
                              void* d_out, int out_size, void* d_ws,
                              size_t ws_size, hipStream_t stream) {
  (void)in_sizes; (void)n_in; (void)out_size; (void)ws_size;
  const float* x  = (const float*)d_in[0];
  const float* Wq = (const float*)d_in[1];
  const float* Wk = (const float*)d_in[2];
  const float* Wv = (const float*)d_in[3];
  float* out = (float*)d_out;

  char* ws = (char*)d_ws;
  const size_t xB = (size_t)BSROWS * DI * sizeof(unsigned short);  // 16 MB
  const size_t wB = (size_t)DI * DI * sizeof(unsigned short);      // 2 MB
  unsigned short* xb    = (unsigned short*)ws;
  unsigned short* wtAll = (unsigned short*)(ws + xB);
  unsigned short* qbf   = (unsigned short*)(ws + xB + 3 * wB);
  unsigned short* kbf   = qbf + (size_t)BSROWS * DI;
  unsigned short* vtbf  = kbf + (size_t)BSROWS * DI;
  // total ws use: 4*16MB + 6MB = 70 MB

  cvt_x_bf16<<<(BSROWS * DI) / (256 * 8), 256, 0, stream>>>(x, xb);
  cvt_w_t<<<dim3(DI / 32, DI / 32, 3), dim3(32, 8), 0, stream>>>(Wq, Wk, Wv,
                                                                 wtAll);
  qkv_gemm<<<dim3(DI / 64, BSROWS / 128, 3), 256, 0, stream>>>(
      xb, wtAll, qbf, kbf, vtbf);
  flash_attn<<<dim3(SS / 16, NB), 256, 0, stream>>>(qbf, kbf, vtbf, out);
}
